// Hybrid_Python_21363167330691
// MI455X (gfx1250) — matrix operand order repaired and checked by reading, not compiled, not run on hardware
//
#include <hip/hip_runtime.h>
#include <math.h>

typedef __attribute__((ext_vector_type(16))) _Float16 v16h;
typedef __attribute__((ext_vector_type(8)))  float    v8f;

// Branchless tanh: native v_tanh_f32 if available (it is on gfx1250), else
// tanh(x) = 1 - 2 / (exp2(2*log2(e)*x) + 1)  -- saturates correctly at +/-inf.
__device__ __forceinline__ float fast_tanh(float x) {
#if __has_builtin(__builtin_amdgcn_tanhf)
    return __builtin_amdgcn_tanhf(x);
#else
    float t = __builtin_amdgcn_exp2f(x * 2.8853900817779268f);
    float r = __builtin_amdgcn_rcpf(t + 1.0f);
    return fmaf(-2.0f, r, 1.0f);
#endif
}

// One wave32 per block; each wave owns 16 trajectories (rows).
// Wave-resident state:
//   - z state: per-lane scalars for row m = lane%16 (replicated across halves)
//   - W2 as 8 f16 WMMA B-tiles (2 K-tiles x 4 N-tiles) in VGPRs
//   - W1/b1 pre-gathered into the exact A-layout element positions
//   - b2 / W3 as per-lane column scalars
__global__ void __launch_bounds__(32)
neural_ode_scan_kernel(const float* __restrict__ z0in,
                       const float* __restrict__ t,
                       const float* __restrict__ prm,
                       const float* __restrict__ W1,
                       const float* __restrict__ b1,
                       const float* __restrict__ W2,
                       const float* __restrict__ b2,
                       const float* __restrict__ W3,
                       const float* __restrict__ b3,
                       float* __restrict__ out,
                       int N, int B)
{
    const int lane = threadIdx.x & 31;
    const int hf   = lane >> 4;     // physical half (0: lanes 0-15, 1: lanes 16-31)
    const int col  = lane & 15;     // N-index within a 16-wide tile; also row m this lane owns
    const int m    = col;
    const int rbase = blockIdx.x * 16;
    const int B2    = 2 * B;

    // ---- scalar ODE params ----
    const float kappa = prm[0];
    const float mass  = prm[2];
    const float coef  = -kappa / mass;
    const float b3s   = b3[0];

    // ---- layer-1 constants gathered into A-layout positions ----
    // A-layout: lane L, element e -> K = kt*32 + 16*(e/8) + 8*(L/16) + (e%8), row M = L%16
    float w1a[2][16], w1b[2][16], bb1[2][16];
#pragma unroll
    for (int kt = 0; kt < 2; ++kt) {
#pragma unroll
        for (int e = 0; e < 16; ++e) {
            int j = kt * 32 + ((e >> 3) << 4) + (hf << 3) + (e & 7);
            w1a[kt][e] = W1[j];        // W1[0][j]
            w1b[kt][e] = W1[64 + j];   // W1[1][j]
            bb1[kt][e] = b1[j];
        }
    }

    // ---- W2 into WMMA B-layout f16 tiles ----
    // B-layout (32x16): lane L, element e -> K = kt*32 + 16*(e/8) + 8*(L/16) + (e%8), N = nt*16 + (L%16)
    v16h Bw2[2][4];
#pragma unroll
    for (int kt = 0; kt < 2; ++kt) {
#pragma unroll
        for (int nt = 0; nt < 4; ++nt) {
            v16h bt;
#pragma unroll
            for (int e = 0; e < 16; ++e) {
                int K = kt * 32 + ((e >> 3) << 4) + (hf << 3) + (e & 7);
                int Nc = nt * 16 + col;
                bt[e] = (_Float16)W2[K * 64 + Nc];
            }
            Bw2[kt][nt] = bt;
        }
    }

    // ---- per-lane column constants for bias2 and W3 ----
    float b2c[4], w3c[4];
#pragma unroll
    for (int nt = 0; nt < 4; ++nt) {
        b2c[nt] = b2[nt * 16 + col];
        w3c[nt] = W3[nt * 16 + col];
    }

    // ---- initial state (row m), replicated across both halves ----
    float z0m = z0in[(rbase + m) * 2 + 0];
    float z1m = z0in[(rbase + m) * 2 + 1];

    // trajectory row 0 = initial state
    __builtin_nontemporal_store(hf ? z1m : z0m,
                                out + (size_t)(rbase + m) * 2 + hf);

    float tcur = t[0];

    for (int i = 0; i < N - 1; ++i) {
        // software-pipelined uniform time load: issued early, used at iteration end
        const float tnext = t[i + 1];

        // ===== layer 1: h1 = tanh(z @ W1 + b1), built directly in A-layout =====
        v16h A0, A1;
#pragma unroll
        for (int e = 0; e < 16; ++e) {
            float h = fmaf(z0m, w1a[0][e], fmaf(z1m, w1b[0][e], bb1[0][e]));
            A0[e] = (_Float16)fast_tanh(h);
        }
#pragma unroll
        for (int e = 0; e < 16; ++e) {
            float h = fmaf(z0m, w1a[1][e], fmaf(z1m, w1b[1][e], bb1[1][e]));
            A1[e] = (_Float16)fast_tanh(h);
        }

        // ===== layer 2: h2 = tanh(h1 @ W2 + b2) via 8x v_wmma_f32_16x16x32_f16 =====
        v8f acc[4];
#pragma unroll
        for (int nt = 0; nt < 4; ++nt) {
            v8f c;
#pragma unroll
            for (int v = 0; v < 8; ++v) c[v] = b2c[nt];   // bias in C (col-only dep)
            c = __builtin_amdgcn_wmma_f32_16x16x32_f16(false, A0, false, Bw2[0][nt],
                                                       (short)0, c, false, false);
            c = __builtin_amdgcn_wmma_f32_16x16x32_f16(false, A1, false, Bw2[1][nt],
                                                       (short)0, c, false, false);
            acc[nt] = c;
        }

        // tanh in place: 32 distinct destination registers -> trans ops can pipeline
#pragma unroll
        for (int nt = 0; nt < 4; ++nt) {
#pragma unroll
            for (int v = 0; v < 8; ++v) {
                acc[nt][v] = fast_tanh(acc[nt][v]);
            }
        }

        // ===== layer 3: nn = h2t @ W3 + b3, consumed in D-layout =====
        // per-lane partial over the 4 column tiles (D: elem v -> row v+8*hf, col nt*16+col)
        float p[8];
#pragma unroll
        for (int v = 0; v < 8; ++v) {
            p[v] = fmaf(acc[0][v], w3c[0],
                   fmaf(acc[1][v], w3c[1],
                   fmaf(acc[2][v], w3c[2], acc[3][v] * w3c[3])));
        }

        // reduce-scatter butterfly over the 16 columns (lane bits 0..3), folding the
        // row-selection (row m = lane%16) into the reduction: at mask-2^k, each lane
        // keeps the register half matching its own bit k and sends the partner's half.
        float s4[4];
#pragma unroll
        for (int j = 0; j < 4; ++j) {              // mask 1, select by lane bit0
            float keep = (lane & 1) ? p[2 * j + 1] : p[2 * j];
            float send = (lane & 1) ? p[2 * j]     : p[2 * j + 1];
            s4[j] = keep + __shfl_xor(send, 1, 32);
        }
        float s2[2];
#pragma unroll
        for (int j = 0; j < 2; ++j) {              // mask 2, select by lane bit1
            float keep = (lane & 2) ? s4[2 * j + 1] : s4[2 * j];
            float send = (lane & 2) ? s4[2 * j]     : s4[2 * j + 1];
            s2[j] = keep + __shfl_xor(send, 2, 32);
        }
        float keep3 = (lane & 4) ? s2[1] : s2[0];  // mask 4, select by lane bit2
        float send3 = (lane & 4) ? s2[0] : s2[1];
        float s1 = keep3 + __shfl_xor(send3, 4, 32);
        s1 += __shfl_xor(s1, 8, 32);               // mask 8: pure column reduce
        // now lane L holds rowSum[(L&7) + 8*bit4(L)]; wanted row-half is bit3(L)
        float o = __shfl_xor(s1, 16, 32);
        bool same = (((lane >> 3) ^ (lane >> 4)) & 1) == 0;
        float nnm = (same ? s1 : o) + b3s;

        // ===== Euler update (simultaneous, as in reference) =====
        const float dt = tnext - tcur;
        tcur = tnext;
        float dz1 = fmaf(coef, z0m, nnm);
        float z0n = fmaf(dt, z1m, z0m);
        float z1n = fmaf(dt, dz1, z1m);
        z0m = z0n;
        z1m = z1n;

        // stream out: half 0 stores z0, half 1 stores z1 for row m
        __builtin_nontemporal_store(hf ? z1n : z0n,
            out + (size_t)(i + 1) * B2 + (size_t)(rbase + m) * 2 + hf);
    }
}

extern "C" void kernel_launch(void* const* d_in, const int* in_sizes, int n_in,
                              void* d_out, int out_size, void* d_ws, size_t ws_size,
                              hipStream_t stream) {
    (void)n_in; (void)out_size; (void)d_ws; (void)ws_size;
    const float* z0 = (const float*)d_in[0];
    const float* t  = (const float*)d_in[1];
    const float* pr = (const float*)d_in[2];
    const float* W1 = (const float*)d_in[3];
    const float* b1 = (const float*)d_in[4];
    const float* W2 = (const float*)d_in[5];
    const float* b2 = (const float*)d_in[6];
    const float* W3 = (const float*)d_in[7];
    const float* b3 = (const float*)d_in[8];
    float* out = (float*)d_out;

    const int B = in_sizes[0] / 2;   // 2048
    const int N = in_sizes[1];       // 10000
    const int blocks = B / 16;       // 128 waves, one wave32 per block

    neural_ode_scan_kernel<<<blocks, 32, 0, stream>>>(
        z0, t, pr, W1, b1, W2, b2, W3, b3, out, N, B);
}
